// TransformerBlock_7146825580646
// MI455X (gfx1250) — matrix operand order repaired and checked by reading, not compiled, not run on hardware
//
#include <hip/hip_runtime.h>

#define DEV __device__ __forceinline__

typedef __bf16 bf16;
typedef __attribute__((ext_vector_type(4)))  __bf16 v4bf;
typedef __attribute__((ext_vector_type(8)))  __bf16 v8bf;
typedef __attribute__((ext_vector_type(16))) __bf16 v16bf;
typedef __attribute__((ext_vector_type(8)))  float  v8f;

static constexpr int Dm   = 1024;   // model dim
static constexpr int SEQ  = 2048;
static constexpr int BAT  = 2;
static constexpr int NH   = 16;
static constexpr int HD   = 64;
static constexpr int ROWS = BAT * SEQ;   // 4096
static constexpr int DFF  = 4096;
static constexpr int LDQKV = 3 * Dm;     // packed QKV row stride

DEV bf16 f2bf(float f) {
  unsigned u = __builtin_bit_cast(unsigned, f);
  unsigned r = u + 0x7FFFu + ((u >> 16) & 1u);
  return __builtin_bit_cast(bf16, (unsigned short)(r >> 16));
}

DEV float gelu_tanh(float x) {
  float x3 = x * x * x;
  return 0.5f * x * (1.0f + tanhf(0.7978845608028654f * (x + 0.044715f * x3)));
}

DEV v8f vzero8() {
  v8f z;
#pragma unroll
  for (int e = 0; e < 8; e++) z[e] = 0.f;
  return z;
}

DEV v16bf cat8(v8bf a, v8bf b) {
  return __builtin_shufflevector(a, b, 0, 1, 2, 3, 4, 5, 6, 7,
                                 8, 9, 10, 11, 12, 13, 14, 15);
}

DEV v8f wmma_bf16(v16bf a, v16bf b, v8f c) {
  return __builtin_amdgcn_wmma_f32_16x16x32_bf16(false, a, false, b,
                                                 (short)0, c, false, false);
}

// Async DMA: global memory -> LDS, 16 bytes per lane. Tracked by ASYNCcnt.
DEV void async_copy_b128(unsigned lds_addr, const void* gaddr) {
  asm volatile("global_load_async_to_lds_b128 %0, %1, off"
               :: "v"(lds_addr), "v"(gaddr) : "memory");
}

// ---------------------------------------------------------------- transpose+cvt
// in: [R][C] fp32 row-major  ->  out: [C][R] bf16 row-major. 64x64 tiles.
__global__ __launch_bounds__(256) void cvt_transpose_kernel(
    const float* __restrict__ in, bf16* __restrict__ out, int R, int C) {
  __shared__ float tile[64][65];
  const int t = threadIdx.x;
  const int r0 = blockIdx.y * 64, c0 = blockIdx.x * 64;
  const int lr = t >> 2;            // 0..63
  const int lc = (t & 3) * 16;      // 0,16,32,48
#pragma unroll
  for (int i = 0; i < 16; i += 4) {
    float4 f = *(const float4*)(in + (size_t)(r0 + lr) * C + c0 + lc + i);
    tile[lr][lc + i + 0] = f.x;
    tile[lr][lc + i + 1] = f.y;
    tile[lr][lc + i + 2] = f.z;
    tile[lr][lc + i + 3] = f.w;
  }
  __syncthreads();
  const int oc = t >> 2;            // column of input -> row of output
  const int orr = (t & 3) * 16;
  v8bf o0, o1;
#pragma unroll
  for (int i = 0; i < 8; i++) o0[i] = f2bf(tile[orr + i][oc]);
#pragma unroll
  for (int i = 0; i < 8; i++) o1[i] = f2bf(tile[orr + 8 + i][oc]);
  bf16* op = out + (size_t)(c0 + oc) * R + r0 + orr;
  *(v8bf*)op = o0;
  *(v8bf*)(op + 8) = o1;
}

// ---------------------------------------------------------------- layernorm
__global__ __launch_bounds__(256) void ln_bf16_kernel(
    const float* __restrict__ x, const float* __restrict__ g,
    const float* __restrict__ s, bf16* __restrict__ out) {
  const int row = blockIdx.x;
  const int t = threadIdx.x;
  const float* xr = x + (size_t)row * Dm;
  __shared__ float red[256];

  float v[4];
  float sum = 0.f;
#pragma unroll
  for (int i = 0; i < 4; i++) { v[i] = xr[t + i * 256]; sum += v[i]; }
  red[t] = sum;
  __syncthreads();
  for (int off = 128; off > 0; off >>= 1) {
    if (t < off) red[t] += red[t + off];
    __syncthreads();
  }
  float mean = red[0] * (1.0f / Dm);
  __syncthreads();
  float sq = 0.f;
#pragma unroll
  for (int i = 0; i < 4; i++) { float d = v[i] - mean; sq += d * d; }
  red[t] = sq;
  __syncthreads();
  for (int off = 128; off > 0; off >>= 1) {
    if (t < off) red[t] += red[t + off];
    __syncthreads();
  }
  float rstd = rsqrtf(red[0] * (1.0f / Dm) + 1e-5f);
  bf16* orow = out + (size_t)row * Dm;
#pragma unroll
  for (int i = 0; i < 4; i++) {
    int c = t + i * 256;
    orow[c] = f2bf(g[c] * ((v[i] - mean) * rstd) + s[c]);
  }
}

// ---------------------------------------------------------------- WMMA GEMM
// C[M,N] = A[M,K] (bf16 row-major) * Bt[N,K]^T (weights pre-transposed),
// fp32 accum. Double-buffered LDS filled by async global->LDS DMA; WMMA burst
// on one buffer overlaps the ASYNCcnt-tracked fill of the other.
// EPI 0: store bf16.  EPI 1: store fp32 = acc + bias + res.  EPI 2: bf16(gelu(acc+bias)).
static constexpr int BM = 128, BN = 128, BK = 64, LDT = 72;

template <int EPI>
__global__ __launch_bounds__(256) void gemm_wmma(
    const bf16* __restrict__ A, const bf16* __restrict__ Bt,
    const float* __restrict__ bias, const float* __restrict__ res,
    void* __restrict__ outp, int M, int N, int K) {
  __shared__ bf16 sA[2][BM * LDT];
  __shared__ bf16 sB[2][BN * LDT];  // [n][k] (Bt rows are B columns)
  const int tid = threadIdx.x;
  const int m0 = blockIdx.y * BM, n0 = blockIdx.x * BN;
  const int wave = tid >> 5, lane = tid & 31, lid = lane & 15, hi = lane >> 4;
  const int wm0 = (wave >> 1) * 32, wn0 = (wave & 1) * 64;

  v8f acc[2][4];
#pragma unroll
  for (int i = 0; i < 2; i++)
#pragma unroll
    for (int j = 0; j < 4; j++) acc[i][j] = vzero8();

  // 8 async b128 ops per thread per tile (4 for A, 4 for B) -> 8 per wave on ASYNCcnt
  auto issue_tile = [&](int buf, int k0) {
#pragma unroll
    for (int c = 0; c < 4; c++) {
      int idx = tid + c * 256;
      int r = idx >> 3, kc = (idx & 7) * 8;
      async_copy_b128((unsigned)(size_t)&sA[buf][r * LDT + kc],
                      (const void*)(A + (size_t)(m0 + r) * K + k0 + kc));
      async_copy_b128((unsigned)(size_t)&sB[buf][r * LDT + kc],
                      (const void*)(Bt + (size_t)(n0 + r) * K + k0 + kc));
    }
  };

  const int T = K / BK;
  issue_tile(0, 0);
  for (int t = 0; t < T; t++) {
    const int cur = t & 1;
    if (t + 1 < T) {
      issue_tile(1 - cur, (t + 1) * BK);
      asm volatile("s_wait_asynccnt 8" ::: "memory");  // tile t landed (in-order)
    } else {
      asm volatile("s_wait_asynccnt 0" ::: "memory");
    }
    __syncthreads();

#pragma unroll
    for (int ks = 0; ks < BK; ks += 32) {
      v16bf af[2];
#pragma unroll
      for (int im = 0; im < 2; im++) {
        const bf16* base = &sA[cur][(wm0 + im * 16 + lid) * LDT + ks + 16 * hi];
        v8bf c0 = *(const v8bf*)base;
        v8bf c1 = *(const v8bf*)(base + 8);
        af[im] = cat8(c0, c1);
      }
#pragma unroll
      for (int in_ = 0; in_ < 4; in_++) {
        v16bf bf_ = *(const v16bf*)&sB[cur][(wn0 + in_ * 16 + lid) * LDT + ks + 16 * hi];
#pragma unroll
        for (int im = 0; im < 2; im++)
          acc[im][in_] = wmma_bf16(af[im], bf_, acc[im][in_]);
      }
    }
    __syncthreads();  // all waves done reading buf[cur] before it is refilled
  }

  // epilogue
#pragma unroll
  for (int in_ = 0; in_ < 4; in_++) {
    int col = n0 + wn0 + in_ * 16 + lid;
    float bval = (EPI != 0) ? bias[col] : 0.f;
#pragma unroll
    for (int im = 0; im < 2; im++) {
#pragma unroll
      for (int r = 0; r < 8; r++) {
        int row = m0 + wm0 + im * 16 + r + 8 * hi;
        float val = acc[im][in_][r];
        size_t idx = (size_t)row * N + col;
        if constexpr (EPI == 0) {
          ((bf16*)outp)[idx] = f2bf(val);
        } else if constexpr (EPI == 1) {
          ((float*)outp)[idx] = val + bval + res[idx];
        } else {
          ((bf16*)outp)[idx] = f2bf(gelu_tanh(val + bval));
        }
      }
    }
  }
}

// ---------------------------------------------------------------- flash attention
// qkv: [B*S, 3*Dm] bf16 packed; q at col h*64, k at Dm+h*64, v at 2*Dm+h*64.
// ctx: [B*S, Dm]. Causal, scale 1/8.
__global__ __launch_bounds__(128) void flash_attn_kernel(
    const bf16* __restrict__ qkv, bf16* __restrict__ ctx) {
  __shared__ bf16 sK[64 * 72];   // [key][d]   (straight copy)
  __shared__ bf16 sVt[64 * 72];  // [d][key]   (transposed)
  __shared__ bf16 sP[4 * 16 * 72];

  const int tid = threadIdx.x;
  const int wave = tid >> 5, lane = tid & 31, lid = lane & 15, hi = lane >> 4;
  const int b = blockIdx.z, h = blockIdx.y, qt = blockIdx.x;
  const size_t rowbase = (size_t)b * SEQ;
  const int qcol = h * HD, kcol = Dm + h * HD, vcol = 2 * Dm + h * HD;
  const int q0 = qt * 64 + wave * 16;
  const float L2E = 1.4426950408889634f;

  // Q A-fragments (d=64 -> two K-steps), loaded straight from global
  v16bf qa[2];
  {
    const bf16* qp = qkv + (rowbase + q0 + lid) * LDQKV + qcol;
#pragma unroll
    for (int d0 = 0; d0 < 2; d0++) {
      v8bf c0 = *(const v8bf*)(qp + d0 * 32 + 16 * hi);
      v8bf c1 = *(const v8bf*)(qp + d0 * 32 + 8 + 16 * hi);
      qa[d0] = cat8(c0, c1);
    }
  }

  float m_i[8], l_i[8];
  v8f o[4];
#pragma unroll
  for (int r = 0; r < 8; r++) { m_i[r] = -1e30f; l_i[r] = 0.f; }
#pragma unroll
  for (int n2 = 0; n2 < 4; n2++) o[n2] = vzero8();

  // V transpose staging geometry: 4 consecutive keys x 8 d per thread
  const int vr4 = (tid >> 3) * 4;   // 0..60
  const int vc8 = (tid & 7) * 8;    // 0..56

  for (int kb = 0; kb <= qt * 64; kb += 64) {
    // stage K (straight b128 copies)
#pragma unroll
    for (int c = 0; c < 4; c++) {
      int chunk = tid + c * 128;  // 0..511
      int r = chunk >> 3;
      int c8 = (chunk & 7) * 8;
      v8bf tk = *(const v8bf*)(qkv + (rowbase + kb + r) * LDQKV + kcol + c8);
      *(v8bf*)&sK[r * 72 + c8] = tk;
    }
    // stage V transposed (packed 4-key runs -> b64 stores)
    {
      v8bf tv[4];
#pragma unroll
      for (int j = 0; j < 4; j++)
        tv[j] = *(const v8bf*)(qkv + (rowbase + kb + vr4 + j) * LDQKV + vcol + vc8);
#pragma unroll
      for (int i = 0; i < 8; i++) {
        v4bf p;
#pragma unroll
        for (int j = 0; j < 4; j++) p[j] = tv[j][i];
        *(v4bf*)&sVt[(vc8 + i) * 72 + vr4] = p;
      }
    }
    __syncthreads();

    // S = Q * K^T  (4 key sub-tiles of 16)
    v8f sc[4];
#pragma unroll
    for (int n = 0; n < 4; n++) {
      v8f z = vzero8();
      v16bf b0 = *(const v16bf*)&sK[(n * 16 + lid) * 72 + 16 * hi];
      v16bf b1 = *(const v16bf*)&sK[(n * 16 + lid) * 72 + 32 + 16 * hi];
      z = wmma_bf16(qa[0], b0, z);
      z = wmma_bf16(qa[1], b1, z);
      sc[n] = z;
    }
    // scale + causal mask
#pragma unroll
    for (int n = 0; n < 4; n++) {
      int kg = kb + n * 16 + lid;
#pragma unroll
      for (int r = 0; r < 8; r++) {
        int qg = q0 + r + 8 * hi;
        float x = sc[n][r] * 0.125f;
        sc[n][r] = (kg > qg) ? -1e30f : x;
      }
    }
    // online softmax (row reductions across the 16 lanes of each half)
#pragma unroll
    for (int r = 0; r < 8; r++) {
      float mx = fmaxf(fmaxf(sc[0][r], sc[1][r]), fmaxf(sc[2][r], sc[3][r]));
      mx = fmaxf(mx, __shfl_xor(mx, 1, 32));
      mx = fmaxf(mx, __shfl_xor(mx, 2, 32));
      mx = fmaxf(mx, __shfl_xor(mx, 4, 32));
      mx = fmaxf(mx, __shfl_xor(mx, 8, 32));
      float mnew = fmaxf(m_i[r], mx);
      float alpha = exp2f((m_i[r] - mnew) * L2E);
      m_i[r] = mnew;
      float sum = 0.f;
#pragma unroll
      for (int n = 0; n < 4; n++) {
        float p = exp2f((sc[n][r] - mnew) * L2E);
        sc[n][r] = p;
        sum += p;
      }
      sum += __shfl_xor(sum, 1, 32);
      sum += __shfl_xor(sum, 2, 32);
      sum += __shfl_xor(sum, 4, 32);
      sum += __shfl_xor(sum, 8, 32);
      l_i[r] = l_i[r] * alpha + sum;
#pragma unroll
      for (int n2 = 0; n2 < 4; n2++) o[n2][r] *= alpha;
    }
    // P: C-layout -> A-layout via per-wave LDS region
    bf16* pw = &sP[wave * 16 * 72];
#pragma unroll
    for (int n = 0; n < 4; n++)
#pragma unroll
      for (int r = 0; r < 8; r++)
        pw[(r + 8 * hi) * 72 + n * 16 + lid] = f2bf(sc[n][r]);

    v16bf pf[2];
#pragma unroll
    for (int kk = 0; kk < 2; kk++) {
      const bf16* base = &pw[lid * 72 + kk * 32 + 16 * hi];
      v8bf c0 = *(const v8bf*)base;
      v8bf c1 = *(const v8bf*)(base + 8);
      pf[kk] = cat8(c0, c1);
    }
    // O += P * V
#pragma unroll
    for (int n2 = 0; n2 < 4; n2++) {
      v16bf vb0 = *(const v16bf*)&sVt[(n2 * 16 + lid) * 72 + 16 * hi];
      v16bf vb1 = *(const v16bf*)&sVt[(n2 * 16 + lid) * 72 + 32 + 16 * hi];
      o[n2] = wmma_bf16(pf[0], vb0, o[n2]);
      o[n2] = wmma_bf16(pf[1], vb1, o[n2]);
    }
    __syncthreads();
  }

  // normalize + store
#pragma unroll
  for (int n2 = 0; n2 < 4; n2++) {
#pragma unroll
    for (int r = 0; r < 8; r++) {
      int qg = q0 + r + 8 * hi;
      float val = o[n2][r] / l_i[r];
      ctx[(rowbase + qg) * Dm + h * HD + n2 * 16 + lid] = f2bf(val);
    }
  }
}

// ---------------------------------------------------------------- launcher
extern "C" void kernel_launch(void* const* d_in, const int* in_sizes, int n_in,
                              void* d_out, int out_size, void* d_ws, size_t ws_size,
                              hipStream_t stream) {
  (void)in_sizes; (void)n_in; (void)out_size; (void)ws_size;
  const float* x  = (const float*)d_in[0];
  const float* wq = (const float*)d_in[1];
  const float* wk = (const float*)d_in[2];
  const float* wv = (const float*)d_in[3];
  const float* wo = (const float*)d_in[4];
  const float* bo = (const float*)d_in[5];
  const float* w1 = (const float*)d_in[6];
  const float* b1 = (const float*)d_in[7];
  const float* w2 = (const float*)d_in[8];
  const float* b2 = (const float*)d_in[9];
  const float* g1 = (const float*)d_in[10];
  const float* s1 = (const float*)d_in[11];
  const float* g2 = (const float*)d_in[12];
  const float* s2 = (const float*)d_in[13];

  char* wsp = (char*)d_ws;
  auto alloc = [&](size_t bytes) -> void* {
    void* p = (void*)wsp;
    wsp += (bytes + 255) & ~(size_t)255;
    return p;
  };
  // all weight buffers are pre-transposed: [N][K] bf16
  bf16* wqkv_t = (bf16*)alloc((size_t)LDQKV * Dm * 2);  // [3072][1024]
  bf16* wo_t   = (bf16*)alloc((size_t)Dm * Dm * 2);     // [1024][1024]
  bf16* w1_t   = (bf16*)alloc((size_t)DFF * Dm * 2);    // [4096][1024]
  bf16* w2_t   = (bf16*)alloc((size_t)Dm * DFF * 2);    // [1024][4096]
  bf16* h_bf   = (bf16*)alloc((size_t)ROWS * Dm * 2);
  bf16* qkv_bf = (bf16*)alloc((size_t)ROWS * LDQKV * 2);
  bf16* ctx_bf = (bf16*)alloc((size_t)ROWS * Dm * 2);
  float* x1    = (float*)alloc((size_t)ROWS * Dm * 4);
  bf16* h2_bf  = (bf16*)alloc((size_t)ROWS * Dm * 2);
  bf16* f1_bf  = (bf16*)alloc((size_t)ROWS * DFF * 2);

  // weight conversion + transpose (redone each call; deterministic)
  dim3 tDD(Dm / 64, Dm / 64);
  cvt_transpose_kernel<<<tDD, 256, 0, stream>>>(wq, wqkv_t, Dm, Dm);
  cvt_transpose_kernel<<<tDD, 256, 0, stream>>>(wk, wqkv_t + (size_t)Dm * Dm, Dm, Dm);
  cvt_transpose_kernel<<<tDD, 256, 0, stream>>>(wv, wqkv_t + (size_t)2 * Dm * Dm, Dm, Dm);
  cvt_transpose_kernel<<<tDD, 256, 0, stream>>>(wo, wo_t, Dm, Dm);
  cvt_transpose_kernel<<<dim3(DFF / 64, Dm / 64), 256, 0, stream>>>(w1, w1_t, Dm, DFF);
  cvt_transpose_kernel<<<dim3(Dm / 64, DFF / 64), 256, 0, stream>>>(w2, w2_t, DFF, Dm);

  // --- attention sub-block ---
  ln_bf16_kernel<<<ROWS, 256, 0, stream>>>(x, g1, s1, h_bf);

  dim3 gQKV(LDQKV / BN, ROWS / BM);  // 24 x 32 (fused Q|K|V projection)
  gemm_wmma<0><<<gQKV, 256, 0, stream>>>(h_bf, wqkv_t, nullptr, nullptr, qkv_bf,
                                         ROWS, LDQKV, Dm);

  dim3 gAttn(SEQ / 64, NH, BAT);  // 32 x 16 x 2
  flash_attn_kernel<<<gAttn, 128, 0, stream>>>(qkv_bf, ctx_bf);

  // x1 = x + ctx @ wo + bo
  dim3 gD(Dm / BN, ROWS / BM);  // 8 x 32
  gemm_wmma<1><<<gD, 256, 0, stream>>>(ctx_bf, wo_t, bo, x, x1, ROWS, Dm, Dm);

  // --- FFN sub-block ---
  ln_bf16_kernel<<<ROWS, 256, 0, stream>>>(x1, g2, s2, h2_bf);

  dim3 gF1(DFF / BN, ROWS / BM);  // 32 x 32
  gemm_wmma<2><<<gF1, 256, 0, stream>>>(h2_bf, w1_t, b1, nullptr, f1_bf,
                                        ROWS, DFF, Dm);

  // out = x1 + gelu(...) @ w2 + b2
  gemm_wmma<1><<<gD, 256, 0, stream>>>(f1_bf, w2_t, b2, x1, (float*)d_out,
                                       ROWS, Dm, DFF);
}
